// LNN_18210661335589
// MI455X (gfx1250) — matrix operand order repaired and checked by reading, not compiled, not run on hardware
//
#include <hip/hip_runtime.h>
#include <hip/hip_bf16.h>

// ---------------------------------------------------------------------------
// LNN dynamics: per-sample Hessian of 2-layer softplus MLP + 16x16 solve.
//   z1 = x@W1+b1, s1 = sigmoid(z1), h1 = softplus(z1)
//   z2 = h1@W2+b2, g2 = W3*sigmoid(z2), d2 = W3*sigmoid'(z2)
//   u  = g2@W2^T,  w = s1*(1-s1)*u,     t  = s1*u
//   jac = t@W1^T
//   M  = (W1 (.) s1)@W2      (32x512 per sample)  <-- dominant: 275 GFLOP total
//   H  = W1 diag(w) W1^T + M diag(d2) M^T
//   A = H[16:,16:], B = H[16:,:16], solve A qdd = jac[:16] - B qd
// GEMMs: v_wmma_f32_16x16x32_bf16 (f32 acc). k4: 2 samples/block (B-frag reuse),
// async global->LDS staging of d2/w rows overlapped with GEMM1.
// ---------------------------------------------------------------------------

typedef __bf16 bf16_t;
typedef __attribute__((ext_vector_type(16))) __bf16 v16bf;
typedef __attribute__((ext_vector_type(8)))  __bf16 v8bf;
typedef __attribute__((ext_vector_type(8)))  float  v8f;

#define BS  16384
#define HID 512
#define DIN 32   // 2*N
#define NN  16

__device__ __forceinline__ v8f zero8() {
  v8f z;
#pragma unroll
  for (int i = 0; i < 8; ++i) z[i] = 0.0f;
  return z;
}

// 32-bit LDS byte offset from a generic pointer to __shared__ memory.
__device__ __forceinline__ unsigned lds_off(const void* p) {
  return (unsigned)(unsigned long long)
      (const __attribute__((address_space(3))) unsigned char*)p;
}

// One 16-byte async global->LDS transfer per active lane (ISA §15.18.3,
// tracked by ASYNCcnt; drained with s_wait_asynccnt).
__device__ __forceinline__ void async_g2l_b128(unsigned lds_byte_off,
                                               const void* gaddr) {
  asm volatile("global_load_async_to_lds_b128 %0, %1, off"
               :
               : "v"(lds_byte_off), "v"(gaddr)
               : "memory");
}
__device__ __forceinline__ void wait_asynccnt0() {
  asm volatile("s_wait_asynccnt 0x0" ::: "memory");
}

// A-fragment (16x32 bf16, M x K): lane L<16 -> row L, K = {half*16+0..15}
// caller passes p = rowbase + kstep*32 + half*16
__device__ __forceinline__ v16bf load_frag_a(const bf16_t* p) {
  v8bf lo = *(const v8bf*)(p);
  v8bf hi = *(const v8bf*)(p + 8);
  return __builtin_shufflevector(lo, hi, 0,1,2,3,4,5,6,7,8,9,10,11,12,13,14,15);
}

// B-fragment (32x16 bf16, K x N): lane holds column (lane&15), 16 contiguous K values.
// caller passes p = colbase(in B^T layout) + kstep*32 + half*16
__device__ __forceinline__ v16bf load_frag_b(const bf16_t* p) {
  return *(const v16bf*)(p);
}

__device__ __forceinline__ v8f wmma_bf16(v16bf a, v16bf b, v8f c) {
  return __builtin_amdgcn_wmma_f32_16x16x32_bf16(false, a, false, b, (short)0, c,
                                                 false, false);
}

__device__ __forceinline__ float sigmoidf_(float z) {
  float e = __expf(-fabsf(z));
  return (z >= 0.0f) ? 1.0f / (1.0f + e) : e / (1.0f + e);
}
__device__ __forceinline__ float softplusf_(float z) {
  return fmaxf(z, 0.0f) + log1pf(__expf(-fabsf(z)));
}

// --------------------------- k0: weight conversion -------------------------
__global__ void k0_convert(const float* __restrict__ W1, const float* __restrict__ W2,
                           bf16_t* __restrict__ W1b, bf16_t* __restrict__ W2b,
                           bf16_t* __restrict__ W2Tb) {
  int idx = blockIdx.x * 256 + threadIdx.x;            // grid covers 512*512
  if (idx < DIN * HID) W1b[idx] = (bf16_t)W1[idx];
  if (idx < HID * HID) {
    float v = W2[idx];
    W2b[idx] = (bf16_t)v;
    int k = idx >> 9, j = idx & (HID - 1);
    W2Tb[j * HID + k] = (bf16_t)v;                     // W2T[j][k] = W2[k][j]
  }
}

// --------------------------- k1: layer 1 (K=32, VALU) ----------------------
__global__ void k1_layer1(const float* __restrict__ x, const float* __restrict__ W1,
                          const float* __restrict__ b1,
                          bf16_t* __restrict__ h1b, bf16_t* __restrict__ s1b) {
  __shared__ float xs[DIN];
  int b = blockIdx.x;
  if (threadIdx.x < DIN) xs[threadIdx.x] = x[(size_t)b * DIN + threadIdx.x];
  __syncthreads();
  for (int n = threadIdx.x; n < HID; n += blockDim.x) {
    float z = b1[n];
#pragma unroll
    for (int i = 0; i < DIN; ++i) z = fmaf(xs[i], W1[i * HID + n], z);
    h1b[(size_t)b * HID + n] = (bf16_t)softplusf_(z);
    s1b[(size_t)b * HID + n] = (bf16_t)sigmoidf_(z);
  }
}

// ------------------ k2: z2 = h1@W2 + b2  ->  g2, d2  (WMMA) ----------------
__global__ __launch_bounds__(256) void k2_layer2(
    const bf16_t* __restrict__ h1b, const bf16_t* __restrict__ W2Tb,
    const float* __restrict__ b2, const float* __restrict__ W3,
    bf16_t* __restrict__ g2b, bf16_t* __restrict__ d2b) {
  int wave = threadIdx.x >> 5, lane = threadIdx.x & 31;
  int half = lane >> 4, l16 = lane & 15;
  size_t mbase = (size_t)blockIdx.x * 16;              // 16 rows per block
  const bf16_t* arow = h1b + (mbase + l16) * HID;
  int ntb = wave * 4;                                  // 8 waves x 4 ntiles = 512 cols
  v8f acc[4];
#pragma unroll
  for (int j = 0; j < 4; ++j) acc[j] = zero8();
  for (int k = 0; k < HID / 32; ++k) {
    v16bf a = load_frag_a(arow + k * 32 + half * 16);
#pragma unroll
    for (int j = 0; j < 4; ++j) {
      int n = (ntb + j) * 16 + l16;
      v16bf bb = load_frag_b(W2Tb + (size_t)n * HID + k * 32 + half * 16);
      acc[j] = wmma_bf16(a, bb, acc[j]);
    }
  }
#pragma unroll
  for (int j = 0; j < 4; ++j) {
    int col = (ntb + j) * 16 + l16;
    float w3 = W3[col], bias = b2[col];
#pragma unroll
    for (int r = 0; r < 8; ++r) {
      size_t row = mbase + r + half * 8;
      float sig = sigmoidf_(acc[j][r] + bias);
      float g2 = w3 * sig;
      g2b[row * HID + col] = (bf16_t)g2;
      d2b[row * HID + col] = (bf16_t)(g2 * (1.0f - sig));
    }
  }
}

// ------------------ k3: u = g2@W2^T -> w, t  (WMMA) ------------------------
__global__ __launch_bounds__(256) void k3_backu(
    const bf16_t* __restrict__ g2b, const bf16_t* __restrict__ W2b,
    const bf16_t* __restrict__ s1b,
    bf16_t* __restrict__ wb, bf16_t* __restrict__ tb) {
  int wave = threadIdx.x >> 5, lane = threadIdx.x & 31;
  int half = lane >> 4, l16 = lane & 15;
  size_t mbase = (size_t)blockIdx.x * 16;
  const bf16_t* arow = g2b + (mbase + l16) * HID;
  int ntb = wave * 4;
  v8f acc[4];
#pragma unroll
  for (int j = 0; j < 4; ++j) acc[j] = zero8();
  for (int k = 0; k < HID / 32; ++k) {
    v16bf a = load_frag_a(arow + k * 32 + half * 16);
#pragma unroll
    for (int j = 0; j < 4; ++j) {
      int n = (ntb + j) * 16 + l16;                    // B col n = W2 row n
      v16bf bb = load_frag_b(W2b + (size_t)n * HID + k * 32 + half * 16);
      acc[j] = wmma_bf16(a, bb, acc[j]);
    }
  }
#pragma unroll
  for (int j = 0; j < 4; ++j) {
    int col = (ntb + j) * 16 + l16;
#pragma unroll
    for (int r = 0; r < 8; ++r) {
      size_t row = mbase + r + half * 8;
      float u = acc[j][r];
      float s1 = (float)s1b[row * HID + col];
      wb[row * HID + col] = (bf16_t)(s1 * (1.0f - s1) * u);
      tb[row * HID + col] = (bf16_t)(s1 * u);
    }
  }
}

// ------------------ k3b: jac_q = t @ W1[:16,:]^T  (WMMA, N=16) -------------
__global__ __launch_bounds__(256) void k3b_jac(
    const bf16_t* __restrict__ tb, const bf16_t* __restrict__ W1b,
    float* __restrict__ jacq) {
  int wave = threadIdx.x >> 5, lane = threadIdx.x & 31;
  int half = lane >> 4, l16 = lane & 15;
  size_t mbase = ((size_t)blockIdx.x * 8 + wave) * 16;
  const bf16_t* arow = tb + (mbase + l16) * HID;
  const bf16_t* brow = W1b + (size_t)l16 * HID;        // B col i = W1 row i (i<16)
  v8f acc = zero8();
  for (int k = 0; k < HID / 32; ++k) {
    v16bf a = load_frag_a(arow + k * 32 + half * 16);
    v16bf bb = load_frag_b(brow + k * 32 + half * 16);
    acc = wmma_bf16(a, bb, acc);
  }
#pragma unroll
  for (int r = 0; r < 8; ++r)
    jacq[(mbase + r + half * 8) * NN + l16] = acc[r];
}

// ------------- k4: per-sample M, Hessian rows, 16x16 solve -----------------
// 2 samples per block: every W2^T B-fragment feeds 2 samples (4 M-tiles),
// halving L2 traffic (8GB -> 4GB). d2/w rows ride in via async LDS loads
// under GEMM1.
__global__ __launch_bounds__(256) void k4_hess(
    const float* __restrict__ x, const bf16_t* __restrict__ W1b,
    const bf16_t* __restrict__ W2Tb, const bf16_t* __restrict__ s1b,
    const bf16_t* __restrict__ d2b, const bf16_t* __restrict__ wb,
    const float* __restrict__ jacq, float* __restrict__ out) {
  __shared__ __align__(64) bf16_t A1[2 * DIN * HID];  // 64KB; reused as A2|A3 blocks
  __shared__ __align__(64) bf16_t Mb[2 * DIN * HID];  // 64KB: M for both samples
  __shared__ __align__(64) bf16_t dwrows[4 * HID];    // [s0:d2][s0:w][s1:d2][s1:w]
  __shared__ float Hs[2 * NN * DIN];                  // M diag(d2) M^T rows
  __shared__ float Hs2[2 * NN * DIN];                 // W1 diag(w) W1^T rows
  __shared__ float Mat[2 * NN * (NN + 1)];

  int b0 = blockIdx.x * 2, tid = threadIdx.x;
  int wave = tid >> 5, lane = tid & 31, half = lane >> 4, l16 = lane & 15;

  // Kick off async staging of d2/w rows (consumed only after GEMM1).
  // 256 lanes x 16B = 4KB. seg: 0=(s0,d2) 1=(s0,w) 2=(s1,d2) 3=(s1,w)
  {
    int seg = tid >> 6, c = tid & 63;                  // c: 16B chunk within row
    const bf16_t* src = ((seg & 1) ? wb : d2b) + ((size_t)(b0 + (seg >> 1))) * HID;
    async_g2l_b128(lds_off(&dwrows[seg * HID + c * 8]), src + c * 8);
  }

  // A1 rows 0..63 = [(W1 (.) s1[b0]) ; (W1 (.) s1[b0+1])]
  for (int idx = tid; idx < 2 * DIN * HID; idx += 256) {
    int r = idx >> 9, kk = idx & (HID - 1);
    int ss = r >> 5, i = r & (DIN - 1);
    A1[idx] = (bf16_t)((float)W1b[i * HID + kk] *
                       (float)s1b[(size_t)(b0 + ss) * HID + kk]);
  }
  __syncthreads();

  // GEMM1: M(64x512) = A1 @ W2. wave -> 4 ntiles x 4 mtiles (16 accs).
  {
    int ntb = wave * 4;
    v8f acc[4][4];
#pragma unroll
    for (int mt = 0; mt < 4; ++mt)
#pragma unroll
      for (int j = 0; j < 4; ++j) acc[mt][j] = zero8();
    for (int k = 0; k < HID / 32; ++k) {
      v16bf bfr[4];
#pragma unroll
      for (int j = 0; j < 4; ++j) {
        int n = (ntb + j) * 16 + l16;
        bfr[j] = load_frag_b(W2Tb + (size_t)n * HID + k * 32 + half * 16);
        if (k + 2 < HID / 32)                           // deepen the pipeline
          __builtin_prefetch(W2Tb + (size_t)n * HID + (k + 2) * 32 + half * 16, 0, 1);
      }
#pragma unroll
      for (int mt = 0; mt < 4; ++mt) {
        v16bf a = load_frag_a(A1 + (mt * 16 + l16) * HID + k * 32 + half * 16);
#pragma unroll
        for (int j = 0; j < 4; ++j) acc[mt][j] = wmma_bf16(a, bfr[j], acc[mt][j]);
      }
    }
#pragma unroll
    for (int mt = 0; mt < 4; ++mt)
#pragma unroll
      for (int j = 0; j < 4; ++j) {
        int col = (ntb + j) * 16 + l16;
#pragma unroll
        for (int r = 0; r < 8; ++r)
          Mb[(mt * 16 + r + half * 8) * HID + col] = (bf16_t)acc[mt][j][r];
      }
  }
  wait_asynccnt0();                                    // d2/w rows have landed
  __syncthreads();

  // Reuse A1, 16-row blocks: [s0:A2][s0:A3][s1:A2][s1:A3]
  //   A2[i][k] = M[ss][16+i][k]*d2[k],  A3[i][k] = W1[16+i][k]*w[k]
  for (int idx = tid; idx < 2 * DIN * HID; idx += 256) {
    int r = idx >> 9, kk = idx & (HID - 1);
    int ss = r >> 5, typ = (r >> 4) & 1, i = r & (NN - 1);
    float scale = (float)dwrows[(ss * 2 + typ) * HID + kk];
    float base = typ ? (float)W1b[(size_t)(NN + i) * HID + kk]
                     : (float)Mb[(ss * DIN + NN + i) * HID + kk];
    A1[idx] = (bf16_t)(base * scale);
  }
  __syncthreads();

  // GEMM2: 8 waves, one 16x16 tile each. ss=wave>>2, isH2=((wave>>1)&1)==0, nt=wave&1
  {
    int ss = wave >> 2, sub = wave & 3;
    int isH2 = (sub < 2) ? 1 : 0, nt = sub & 1;
    const bf16_t* arow = A1 + ((ss * DIN) + (isH2 ? 0 : NN) + l16) * HID;
    v8f acc = zero8();
    for (int k = 0; k < HID / 32; ++k) {
      v16bf a = load_frag_a(arow + k * 32 + half * 16);
      int m = nt * 16 + l16;                           // B col m = M row m / W1 row m
      const bf16_t* bcol = isH2 ? (Mb + (size_t)(ss * DIN + m) * HID)
                                : (W1b + (size_t)m * HID);
      v16bf bb = load_frag_b(bcol + k * 32 + half * 16);
      acc = wmma_bf16(a, bb, acc);
    }
    float* dst = (isH2 ? Hs : Hs2) + ss * NN * DIN;
#pragma unroll
    for (int r = 0; r < 8; ++r)
      dst[(r + half * 8) * DIN + nt * 16 + l16] = acc[r];
  }
  __syncthreads();
  for (int idx = tid; idx < 2 * NN * DIN; idx += 256) Hs[idx] += Hs2[idx];
  __syncthreads();

  // 16x16 solve with partial pivoting: A qdd = jac_q - Bm qd  (one thread/sample)
  if (tid < 2) {
    int b = b0 + tid;
    const float* H = Hs + tid * NN * DIN;
    float* Mt = Mat + tid * NN * (NN + 1);
    float qd[NN];
#pragma unroll
    for (int j = 0; j < NN; ++j) qd[j] = x[(size_t)b * DIN + NN + j];
    for (int i = 0; i < NN; ++i) {
      float r = jacq[(size_t)b * NN + i];
      for (int j = 0; j < NN; ++j) r -= H[i * DIN + j] * qd[j];    // Bm = H[16:,:16]
      Mt[i * 17 + 16] = r;
      for (int j = 0; j < NN; ++j) Mt[i * 17 + j] = H[i * DIN + NN + j];  // A
    }
    for (int c = 0; c < NN; ++c) {
      int piv = c; float best = fabsf(Mt[c * 17 + c]);
      for (int r2 = c + 1; r2 < NN; ++r2) {
        float v = fabsf(Mt[r2 * 17 + c]);
        if (v > best) { best = v; piv = r2; }
      }
      if (piv != c)
        for (int j = c; j < 17; ++j) {
          float t = Mt[c * 17 + j]; Mt[c * 17 + j] = Mt[piv * 17 + j];
          Mt[piv * 17 + j] = t;
        }
      float p = Mt[c * 17 + c];
      if (fabsf(p) < 1e-30f) p = (p >= 0.0f ? 1e-30f : -1e-30f);
      float inv = 1.0f / p;
      for (int j = c; j < 17; ++j) Mt[c * 17 + j] *= inv;
      for (int r2 = 0; r2 < NN; ++r2) {
        if (r2 == c) continue;
        float f = Mt[r2 * 17 + c];
        for (int j = c; j < 17; ++j) Mt[r2 * 17 + j] -= f * Mt[c * 17 + j];
      }
    }
    for (int j = 0; j < NN; ++j) {
      out[(size_t)b * DIN + j] = qd[j];
      out[(size_t)b * DIN + NN + j] = Mt[j * 17 + 16];
    }
  }
}

// ---------------------------------------------------------------------------
extern "C" void kernel_launch(void* const* d_in, const int* in_sizes, int n_in,
                              void* d_out, int out_size, void* d_ws, size_t ws_size,
                              hipStream_t stream) {
  const float* x  = (const float*)d_in[0];
  const float* W1 = (const float*)d_in[1];
  const float* b1 = (const float*)d_in[2];
  const float* W2 = (const float*)d_in[3];
  const float* b2 = (const float*)d_in[4];
  const float* W3 = (const float*)d_in[5];
  float* out = (float*)d_out;

  char* ws = (char*)d_ws;
  const size_t SB = (size_t)BS * HID * sizeof(bf16_t);   // 16 MB per bf16 field
  bf16_t* h1b = (bf16_t*)(ws + 0 * SB);
  bf16_t* s1b = (bf16_t*)(ws + 1 * SB);
  bf16_t* g2b = (bf16_t*)(ws + 2 * SB);
  bf16_t* d2b = (bf16_t*)(ws + 3 * SB);
  bf16_t* wb  = (bf16_t*)(ws + 4 * SB);
  bf16_t* tb  = (bf16_t*)(ws + 5 * SB);
  float*  jacq = (float*)(ws + 6 * SB);                  // 1 MB
  bf16_t* W1b  = (bf16_t*)(ws + 6 * SB + (1u << 20));
  bf16_t* W2b  = W1b + DIN * HID;
  bf16_t* W2Tb = W2b + HID * HID;

  k0_convert<<<(HID * HID) / 256, 256, 0, stream>>>(W1, W2, W1b, W2b, W2Tb);
  k1_layer1<<<BS, 128, 0, stream>>>(x, W1, b1, h1b, s1b);
  k2_layer2<<<BS / 16, 256, 0, stream>>>(h1b, W2Tb, b2, W3, g2b, d2b);
  k3_backu<<<BS / 16, 256, 0, stream>>>(g2b, W2b, s1b, wb, tb);
  k3b_jac<<<BS / 128, 256, 0, stream>>>(tb, W1b, jacq);
  k4_hess<<<BS / 2, 256, 0, stream>>>(x, W1b, W2Tb, s1b, d2b, wb, jacq, out);
}
